// Head_67937792688536
// MI455X (gfx1250) — matrix operand order repaired and checked by reading, not compiled, not run on hardware
//
#include <hip/hip_runtime.h>
#include <hip/hip_bf16.h>
#include <math.h>

typedef __attribute__((ext_vector_type(16))) _Float16 v16h;
typedef __attribute__((ext_vector_type(8)))  _Float16 h8v;
typedef __attribute__((ext_vector_type(8)))  float    v8f;

#define WMMA_F32_F16(a, b, c) \
  __builtin_amdgcn_wmma_f32_16x16x32_f16(false, (a), false, (b), (short)0, (c), false, false)

static constexpr int Bn = 8;     // batch
static constexpr int Tn = 2048;  // sequence length
static constexpr int Cn = 1024;  // channels
static constexpr int HS = 64;    // head size

// Join two contiguous 8xf16 vectors into one WMMA 16xf16 operand register tuple.
__device__ __forceinline__ v16h join16(h8v lo, h8v hi) {
  return __builtin_shufflevector(lo, hi, 0, 1, 2, 3, 4, 5, 6, 7,
                                 8, 9, 10, 11, 12, 13, 14, 15);
}

// ---------------------------------------------------------------------------
// Kernel 0: transpose + convert the three (C x HS) fp32 weights into
// Wt[which][n][k] f16 so the projection B-operand (VGPR j: K = 16*half + 2j)
// is 16 contiguous f16 = one 32-byte vector load.
// ---------------------------------------------------------------------------
__global__ __launch_bounds__(256) void conv_w_kernel(
    const float* __restrict__ Wq, const float* __restrict__ Wk,
    const float* __restrict__ Wv, _Float16* __restrict__ Wt)
{
  const int which = blockIdx.y;
  const float* W = (which == 0) ? Wq : (which == 1) ? Wk : Wv;
  _Float16* O = Wt + (size_t)which * HS * Cn;
  const int idx = blockIdx.x * 256 + threadIdx.x;   // 0 .. HS*Cn-1
  const int n = idx >> 10;                          // head col
  const int k = idx & (Cn - 1);                     // channel
  O[idx] = (_Float16)W[(size_t)k * HS + n];
}

// ---------------------------------------------------------------------------
// Kernel 1: QKV projection. One wave computes one 16x64 tile of one of
// {Q, K, V} with v_wmma_f32_16x16x32_f16, f32 accumulation.
// A: X fp32 via float4 loads + packed cvt. B: Wt f16 via 32B vector loads.
// Q, K stored f16 row-major; V stored transposed Vt[b][h][t].
// ---------------------------------------------------------------------------
__global__ __launch_bounds__(32) void qkv_proj_kernel(
    const float* __restrict__ X, const _Float16* __restrict__ Wt,
    _Float16* __restrict__ Q, _Float16* __restrict__ Kf,
    _Float16* __restrict__ Vt)
{
  const int mtile = blockIdx.x;
  const int which = blockIdx.y;                    // 0=Q 1=K 2=V
  const _Float16* Wb = Wt + (size_t)which * HS * Cn;

  const int lane = threadIdx.x & 31;
  const int lm = lane & 15;
  const int hi = lane >> 4;
  const int m  = mtile * 16 + lm;

  v8f acc[4] = {v8f{}, v8f{}, v8f{}, v8f{}};

  for (int kc = 0; kc < Cn / 32; ++kc) {
    const int kb = kc * 32;
    // A 16x32 f16: elements 0..15 = K kb+16*hi+t
    const float* xp = X + (size_t)m * Cn + kb + 16 * hi;
    const float4 f0 = *(const float4*)(xp + 0);
    const float4 f1 = *(const float4*)(xp + 4);
    const float4 f2 = *(const float4*)(xp + 8);
    const float4 f3 = *(const float4*)(xp + 12);
    v16h a;
    a[0]  = (_Float16)f0.x; a[1]  = (_Float16)f0.y; a[2]  = (_Float16)f0.z; a[3]  = (_Float16)f0.w;
    a[4]  = (_Float16)f1.x; a[5]  = (_Float16)f1.y; a[6]  = (_Float16)f1.z; a[7]  = (_Float16)f1.w;
    a[8]  = (_Float16)f2.x; a[9]  = (_Float16)f2.y; a[10] = (_Float16)f2.z; a[11] = (_Float16)f2.w;
    a[12] = (_Float16)f3.x; a[13] = (_Float16)f3.y; a[14] = (_Float16)f3.z; a[15] = (_Float16)f3.w;
#pragma unroll
    for (int nt = 0; nt < 4; ++nt) {
      const int n = nt * 16 + lm;
      // B 32x16 f16: 16 contiguous f16 at Wt[n][kb + 16*hi]
      const v16h bmat = *(const v16h*)(Wb + (size_t)n * Cn + kb + hi * 16);
      acc[nt] = WMMA_F32_F16(a, bmat, acc[nt]);
    }
  }

  // C/D layout: lane holds (row = r + 8*half, col = lm) in VGPR r.
  if (which == 2) {
#pragma unroll
    for (int nt = 0; nt < 4; ++nt)
#pragma unroll
      for (int r = 0; r < 8; ++r) {
        const int row = mtile * 16 + r + 8 * hi;     // global b*T + t
        const int col = nt * 16 + lm;
        const int bb = row / Tn, tt = row % Tn;
        Vt[((size_t)bb * HS + col) * Tn + tt] = (_Float16)acc[nt][r];
      }
  } else {
    _Float16* Out = (which == 0) ? Q : Kf;
#pragma unroll
    for (int nt = 0; nt < 4; ++nt)
#pragma unroll
      for (int r = 0; r < 8; ++r) {
        const int row = mtile * 16 + r + 8 * hi;
        const int col = nt * 16 + lm;
        Out[(size_t)row * HS + col] = (_Float16)acc[nt][r];
      }
  }
}

// ---------------------------------------------------------------------------
// Kernel 2: flash attention, one wave per 16-query tile, 32-key blocks,
// online softmax in exp2 domain, causal mask only on the diagonal block.
// ---------------------------------------------------------------------------
__global__ __launch_bounds__(32) void flash_attn_kernel(
    const _Float16* __restrict__ Q, const _Float16* __restrict__ Kf,
    const _Float16* __restrict__ Vt, float* __restrict__ O)
{
  __shared__ __align__(32) _Float16 pbuf[16 * 32];   // one wave per block

  const int qtile = blockIdx.x;
  const int b     = blockIdx.y;
  const int lane  = threadIdx.x & 31;
  const int lm = lane & 15, hi = lane >> 4;

  const _Float16* Qb = Q  + (size_t)b * Tn * HS;
  const _Float16* Kb = Kf + (size_t)b * Tn * HS;
  const _Float16* Vb = Vt + (size_t)b * HS * Tn;

  // Q tile in A layout: chunk c elements 0..15 at [mrow][c*32 + 16*hi]
  // -> two 16B loads each.
  const int mrowQ = qtile * 16 + lm;
  v16h aq[2];
#pragma unroll
  for (int c = 0; c < 2; ++c) {
    const _Float16* qp = Qb + (size_t)mrowQ * HS + c * 32 + 16 * hi;
    aq[c] = join16(*(const h8v*)qp, *(const h8v*)(qp + 8));
  }

  v8f o0{}, o1{}, o2{}, o3{};
  v8f mrun, lrun;
#pragma unroll
  for (int r = 0; r < 8; ++r) { mrun[r] = -__builtin_inff(); lrun[r] = 0.0f; }

  const int   nkb = qtile / 2 + 1;             // 32-key blocks in causal band
  const float sl2 = 8.0f * 1.4426950408889634f; // sqrt(HS) * log2(e)

  for (int kbi = 0; kbi < nkb; ++kbi) {
    const int  k0     = kbi * 32;
    const bool masked = (kbi == nkb - 1);      // only diagonal block needs mask

    if (kbi + 1 < nkb) {
      __builtin_prefetch(Kb + (size_t)(k0 + 32) * HS, 0, 1);          // K block
      __builtin_prefetch(Vb + (size_t)lm * Tn + (k0 + 32), 0, 1);     // V block
    }

    // ---- S = Q x K^T : two 16x16 C tiles over 32 keys ----
    // B operand = 16 contiguous f16 at K[key][hi*16 (+32)]  (one 32B load)
    v8f s0{}, s1{};
#pragma unroll
    for (int s = 0; s < 2; ++s) {
      const _Float16* kp = Kb + (size_t)(k0 + s * 16 + lm) * HS + hi * 16;
      const v16h bk0 = *(const v16h*)(kp);        // head dims 0..31
      const v16h bk1 = *(const v16h*)(kp + 32);   // head dims 32..63
      if (s == 0) { s0 = WMMA_F32_F16(aq[0], bk0, s0); s0 = WMMA_F32_F16(aq[1], bk1, s0); }
      else        { s1 = WMMA_F32_F16(aq[0], bk0, s1); s1 = WMMA_F32_F16(aq[1], bk1, s1); }
    }

    // ---- online softmax (exp2 domain), fused per-row ----
#pragma unroll
    for (int r = 0; r < 8; ++r) {
      float x0 = s0[r] * sl2;
      float x1 = s1[r] * sl2;
      if (masked) {
        const int mrow = qtile * 16 + r + 8 * hi;
        if (k0 + lm      > mrow) x0 = -__builtin_inff();
        if (k0 + 16 + lm > mrow) x1 = -__builtin_inff();
      }
      float rm = fmaxf(x0, x1);
      rm = fmaxf(rm, __shfl_xor(rm, 1));
      rm = fmaxf(rm, __shfl_xor(rm, 2));
      rm = fmaxf(rm, __shfl_xor(rm, 4));
      rm = fmaxf(rm, __shfl_xor(rm, 8));
      const float mnew  = fmaxf(mrun[r], rm);
      const float alpha = exp2f(mrun[r] - mnew);   // 0 on first block
      const float e0 = exp2f(x0 - mnew);
      const float e1 = exp2f(x1 - mnew);
      float rs = e0 + e1;
      rs += __shfl_xor(rs, 1);
      rs += __shfl_xor(rs, 2);
      rs += __shfl_xor(rs, 4);
      rs += __shfl_xor(rs, 8);
      lrun[r] = lrun[r] * alpha + rs;
      mrun[r] = mnew;
      o0[r] *= alpha; o1[r] *= alpha; o2[r] *= alpha; o3[r] *= alpha;
      // stage P (row-major 16x32 f16) for C->A relayout
      const int row = r + 8 * hi;
      pbuf[row * 32 + lm]      = (_Float16)e0;
      pbuf[row * 32 + 16 + lm] = (_Float16)e1;
    }

    __builtin_amdgcn_wave_barrier();
    asm volatile("s_wait_dscnt 0" ::: "memory");   // cross-lane LDS RAW

    // P in A layout: two contiguous 16B chunks per lane -> ds_load_b128 x2
    const _Float16* pp = pbuf + lm * 32 + 16 * hi;
    const v16h ap = join16(*(const h8v*)pp, *(const h8v*)(pp + 8));
    __builtin_amdgcn_wave_barrier();

    // ---- O += P x V : B operand = 16 contiguous f16 in Vt[n][k0 + hi*16] ----
#pragma unroll
    for (int nt = 0; nt < 4; ++nt) {
      const int n = nt * 16 + lm;
      const v16h bv = *(const v16h*)(Vb + (size_t)n * Tn + k0 + hi * 16);
      if      (nt == 0) o0 = WMMA_F32_F16(ap, bv, o0);
      else if (nt == 1) o1 = WMMA_F32_F16(ap, bv, o1);
      else if (nt == 2) o2 = WMMA_F32_F16(ap, bv, o2);
      else              o3 = WMMA_F32_F16(ap, bv, o3);
    }
  }

  // ---- epilogue: normalize and store fp32 ----
  float* Ob = O + (size_t)b * Tn * HS;
#pragma unroll
  for (int r = 0; r < 8; ++r) {
    const int row = qtile * 16 + r + 8 * hi;
    const float inv = 1.0f / lrun[r];
    Ob[(size_t)row * HS +  0 + lm] = o0[r] * inv;
    Ob[(size_t)row * HS + 16 + lm] = o1[r] * inv;
    Ob[(size_t)row * HS + 32 + lm] = o2[r] * inv;
    Ob[(size_t)row * HS + 48 + lm] = o3[r] * inv;
  }
}

// ---------------------------------------------------------------------------
extern "C" void kernel_launch(void* const* d_in, const int* in_sizes, int n_in,
                              void* d_out, int out_size, void* d_ws, size_t ws_size,
                              hipStream_t stream) {
  (void)in_sizes; (void)n_in; (void)out_size; (void)ws_size;

  const float* X  = (const float*)d_in[0];   // (B, T, C)
  const float* Wq = (const float*)d_in[1];   // (C, HS)
  const float* Wk = (const float*)d_in[2];
  const float* Wv = (const float*)d_in[3];
  float* Out = (float*)d_out;                // (B, T, HS)

  // f16 scratch: Q (2MB) + K (2MB) + Vt (2MB) + Wt (384KB)
  const size_t elems = (size_t)Bn * Tn * HS;
  _Float16* Qh = (_Float16*)d_ws;
  _Float16* Kh = Qh + elems;
  _Float16* Vh = Kh + elems;
  _Float16* Wt = Vh + elems;

  conv_w_kernel<<<dim3((HS * Cn) / 256, 3), 256, 0, stream>>>(Wq, Wk, Wv, Wt);
  qkv_proj_kernel<<<dim3((Bn * Tn) / 16, 3), 32, 0, stream>>>(X, Wt, Qh, Kh, Vh);
  flash_attn_kernel<<<dim3(Tn / 16, Bn), 32, 0, stream>>>(Qh, Kh, Vh, Out);
}
